// NeighborWiseAttention_38019050504716
// MI455X (gfx1250) — matrix operand order repaired and checked by reading, not compiled, not run on hardware
//
#include <hip/hip_runtime.h>
#include <hip/hip_bf16.h>

// ---------------------------------------------------------------------------
// Fused neighbor-wise attention for MI455X (gfx1250), bf16 WMMA + f32 accum.
// One workgroup (8 wave32) per batch element; all intermediates in LDS.
// ---------------------------------------------------------------------------

typedef __bf16 bf16t;
typedef __attribute__((ext_vector_type(16))) __bf16 v16bf;
typedef __attribute__((ext_vector_type(8)))  __bf16 v8bf;
typedef __attribute__((ext_vector_type(8)))  float  v8f;

#define NB      4096
#define NNEI    128
#define EMBED   64
#define HIDDEN  128
#define NLAYER  2
#define ATTNW_SHIFT 20.0f
#define LN_EPS      1e-5f
#define SCALING     0.08838834764831845f   // HIDDEN^-0.5

// LDS strides (elements) — padded so every WMMA fetch is 16B aligned and
// row bases rotate across the 64 LDS banks.
#define XS_LD  68     // f32 x (residual)
#define XB_LD  72     // bf16 x
#define QS_LD  136    // bf16 q / kT / v / attn / o

// LDS byte offsets
#define OFF_XS   0                        // 128*68*4  = 34816
#define OFF_XB   34816                    // 128*72*2  = 18432
#define OFF_QS   53248                    // 128*136*2 = 34816
#define OFF_KT   88064
#define OFF_VS   122880
#define OFF_AT   157696
#define OFF_OS   192512
#define OFF_SW   227328                   // 128*4
#define OFF_MK   227840                   // 128*4
#define OFF_R    228352                   // 128*4*4
#define SMEM_BYTES 230400

__device__ __forceinline__ v8f wmma_bf16(v16bf a, v16bf b, v8f c) {
  // D(f32 16x16) = A(16x32 bf16) * B(32x16 bf16) + C
  return __builtin_amdgcn_wmma_f32_16x16x32_bf16(
      false, a, false, b, (short)0, c, false, false);
}

// A-operand fetch: per lane 8 bf16 at p, 8 bf16 at p+16 (K and K+16 chunks).
__device__ __forceinline__ v16bf load16_gap(const bf16t* p) {
  v8bf lo = *(const v8bf*)p;
  v8bf hi = *(const v8bf*)(p + 16);
  v16bf r;
#pragma unroll
  for (int i = 0; i < 8; ++i) { r[i] = lo[i]; r[i + 8] = hi[i]; }
  return r;
}

// B-operand fetch: per lane column N, rows K and K+16 chunks (stride ld).
__device__ __forceinline__ v16bf load16_seq(const bf16t* p, int ld) {
  v8bf lo, hi;
#pragma unroll
  for (int i = 0; i < 8; ++i) { lo[i] = p[i * ld]; hi[i] = p[(16 + i) * ld]; }
  v16bf r;
#pragma unroll
  for (int i = 0; i < 8; ++i) { r[i] = lo[i]; r[i + 8] = hi[i]; }
  return r;
}

// Reductions across each 16-lane half of a wave32 (rows live in halves).
__device__ __forceinline__ float redsum16(float v) {
  v += __shfl_xor(v, 1, 32);
  v += __shfl_xor(v, 2, 32);
  v += __shfl_xor(v, 4, 32);
  v += __shfl_xor(v, 8, 32);
  return v;
}
__device__ __forceinline__ float redmax16(float v) {
  v = fmaxf(v, __shfl_xor(v, 1, 32));
  v = fmaxf(v, __shfl_xor(v, 2, 32));
  v = fmaxf(v, __shfl_xor(v, 4, 32));
  v = fmaxf(v, __shfl_xor(v, 8, 32));
  return v;
}

// ---------------------------------------------------------------------------
// Prepass: convert weights f32 -> bf16 once (resident in 192MB L2 afterwards).
// ---------------------------------------------------------------------------
__global__ void convert_weights_kernel(const float* __restrict__ in_w,
                                       const float* __restrict__ out_w,
                                       bf16t* __restrict__ wi,
                                       bf16t* __restrict__ wo) {
  int i = blockIdx.x * 256 + threadIdx.x;
  const int n1 = NLAYER * EMBED * 3 * HIDDEN;   // 49152
  const int n2 = NLAYER * HIDDEN * EMBED;       // 16384
  if (i < n1) wi[i] = (bf16t)in_w[i];
  if (i < n2) wo[i] = (bf16t)out_w[i];
}

// ---------------------------------------------------------------------------
// Main fused kernel
// ---------------------------------------------------------------------------
__global__ __launch_bounds__(256)
void attn_fused_kernel(const float* __restrict__ G,
                       const unsigned char* __restrict__ nmask,  // jnp bool = 1B
                       const float* __restrict__ rvec,
                       const float* __restrict__ sw,
                       const float* __restrict__ in_b,
                       const float* __restrict__ out_b,
                       const float* __restrict__ ln_g,
                       const float* __restrict__ ln_b,
                       const bf16t* __restrict__ Wi_all,
                       const bf16t* __restrict__ Wo_all,
                       float* __restrict__ out) {
  extern __shared__ char smem[];
  float* xs  = (float*)(smem + OFF_XS);   // [128][68]  residual / x (f32)
  bf16t* xb  = (bf16t*)(smem + OFF_XB);   // [128][72]  x (bf16, A operand)
  bf16t* qs  = (bf16t*)(smem + OFF_QS);   // [128][136] q row-major
  bf16t* kT  = (bf16t*)(smem + OFF_KT);   // [128][136] k transposed: kT[h][m]
  bf16t* vs  = (bf16t*)(smem + OFF_VS);   // [128][136] v row-major
  bf16t* at  = (bf16t*)(smem + OFF_AT);   // [128][136] attn row-major
  bf16t* os  = (bf16t*)(smem + OFF_OS);   // [128][136] o row-major
  float* swL = (float*)(smem + OFF_SW);
  float* mkL = (float*)(smem + OFF_MK);
  float* rL  = (float*)(smem + OFF_R);

  const int tid  = threadIdx.x;
  const int lane = tid & 31;
  const int wave = tid >> 5;          // 0..7, owns rows 16w..16w+15
  const int lrow = lane & 15;         // M (A) or N (B/C) index within tile
  const int lhi  = (lane >> 4) & 1;   // half-wave select
  const int row0 = wave * 16;
  const size_t bidx = blockIdx.x;

  // ---- stage per-batch tensors into LDS --------------------------------
  const float* Gp = G + bidx * (size_t)(NNEI * EMBED);
#pragma unroll 4
  for (int i = tid; i < NNEI * EMBED; i += 256) {
    float v = Gp[i];
    int rI = i >> 6, cI = i & 63;
    xs[rI * XS_LD + cI] = v;
    xb[rI * XB_LD + cI] = (bf16t)v;
  }
  if (tid < NNEI) {
    swL[tid] = sw[bidx * NNEI + tid];
    mkL[tid] = nmask[bidx * NNEI + tid] ? 1.0f : 0.0f;
    size_t rb = (bidx * NNEI + tid) * 3;
    rL[tid * 4 + 0] = rvec[rb + 0];
    rL[tid * 4 + 1] = rvec[rb + 1];
    rL[tid * 4 + 2] = rvec[rb + 2];
    rL[tid * 4 + 3] = 0.0f;
  }
  __syncthreads();

  for (int layer = 0; layer < NLAYER; ++layer) {
    const bf16t* Wi = Wi_all + layer * (EMBED * 3 * HIDDEN);
    const bf16t* Wo = Wo_all + layer * (HIDDEN * EMBED);
    const float* bi = in_b  + layer * (3 * HIDDEN);
    const float* bo = out_b + layer * EMBED;
    const float* lg = ln_g  + layer * EMBED;
    const float* lb = ln_b  + layer * EMBED;

    // ===== GEMM1: qkv = x @ w_in + b_in ; row-wise l2norm ================
    {
      const bf16t* arow = xb + (row0 + lrow) * XB_LD + lhi * 8;
      v16bf a0 = load16_gap(arow);        // K = 0..31
      v16bf a1 = load16_gap(arow + 32);   // K = 32..63
      for (int grp = 0; grp < 3; ++grp) { // q, k, v (128 cols each)
        const int colbase = grp * HIDDEN;
        v8f c[8] = {};
#pragma unroll
        for (int nt = 0; nt < 8; ++nt) {
          const bf16t* bp = Wi + (lhi * 8) * (3 * HIDDEN) + colbase + nt * 16 + lrow;
          v16bf b0 = load16_seq(bp, 3 * HIDDEN);                        // K rows 0..31
          v16bf b1 = load16_seq(bp + 32 * (3 * HIDDEN), 3 * HIDDEN);    // K rows 32..63
          c[nt] = wmma_bf16(a0, b0, c[nt]);
          c[nt] = wmma_bf16(a1, b1, c[nt]);
        }
#pragma unroll
        for (int nt = 0; nt < 8; ++nt) {
          float bias = bi[colbase + nt * 16 + lrow];
#pragma unroll
          for (int g = 0; g < 8; ++g) c[nt][g] += bias;
        }
#pragma unroll
        for (int g = 0; g < 8; ++g) {
          float ss = 0.f;
#pragma unroll
          for (int nt = 0; nt < 8; ++nt) ss += c[nt][g] * c[nt][g];
          ss = redsum16(ss);
          float scale = 1.0f / fmaxf(sqrtf(ss), 1e-12f);
          if (grp == 0) scale *= SCALING;
          int rowg = row0 + g + 8 * lhi;
#pragma unroll
          for (int nt = 0; nt < 8; ++nt) {
            int cc = nt * 16 + lrow;
            bf16t hv = (bf16t)(c[nt][g] * scale);
            if (grp == 0)      qs[rowg * QS_LD + cc] = hv;
            else if (grp == 1) kT[cc * QS_LD + rowg] = hv;   // transpose k
            else               vs[rowg * QS_LD + cc] = hv;
          }
        }
      }
    }
    __syncthreads();

    // ===== GEMM2: scores = q @ k^T ; gate + softmax + mask + angular =====
    {
      v8f c[8] = {};
#pragma unroll
      for (int s = 0; s < 4; ++s) {       // K = hidden, 4 steps of 32
        v16bf a = load16_gap(qs + (row0 + lrow) * QS_LD + s * 32 + lhi * 8);
#pragma unroll
        for (int nt = 0; nt < 8; ++nt) {
          v16bf b = load16_seq(kT + (s * 32 + lhi * 8) * QS_LD + nt * 16 + lrow, QS_LD);
          c[nt] = wmma_bf16(a, b, c[nt]);
        }
      }
      float swm[8], rm0[8], rm1[8], rm2[8];
#pragma unroll
      for (int nt = 0; nt < 8; ++nt) {
        int m = nt * 16 + lrow;
        swm[nt] = swL[m];
        rm0[nt] = rL[m * 4 + 0];
        rm1[nt] = rL[m * 4 + 1];
        rm2[nt] = rL[m * 4 + 2];
      }
#pragma unroll
      for (int g = 0; g < 8; ++g) {
        int rowg = row0 + g + 8 * lhi;
        float swn = swL[rowg];
        float mk  = mkL[rowg];
        float rn0 = rL[rowg * 4 + 0];
        float rn1 = rL[rowg * 4 + 1];
        float rn2 = rL[rowg * 4 + 2];
        float t[8];
        float mx = -3.4e38f;
#pragma unroll
        for (int nt = 0; nt < 8; ++nt) {
          t[nt] = (c[nt][g] + ATTNW_SHIFT) * (swn * swm[nt]) - ATTNW_SHIFT;
          mx = fmaxf(mx, t[nt]);
        }
        mx = redmax16(mx);
        float sum = 0.f;
#pragma unroll
        for (int nt = 0; nt < 8; ++nt) { t[nt] = __expf(t[nt] - mx); sum += t[nt]; }
        sum = redsum16(sum);
        float inv = 1.0f / sum;
#pragma unroll
        for (int nt = 0; nt < 8; ++nt) {
          float ang = rn0 * rm0[nt] + rn1 * rm1[nt] + rn2 * rm2[nt];
          float av  = t[nt] * inv * mk * (swn * swm[nt]) * ang;
          at[rowg * QS_LD + nt * 16 + lrow] = (bf16t)av;
        }
      }
    }
    // attn rows + o rows are wave-local; LDS is in-order per wave -> no barrier

    // ===== GEMM2b: o = attn @ v ==========================================
    {
      v8f oc[8] = {};
#pragma unroll
      for (int s = 0; s < 4; ++s) {       // K = nnei
        v16bf a = load16_gap(at + (row0 + lrow) * QS_LD + s * 32 + lhi * 8);
#pragma unroll
        for (int nt = 0; nt < 8; ++nt) {
          v16bf b = load16_seq(vs + (s * 32 + lhi * 8) * QS_LD + nt * 16 + lrow, QS_LD);
          oc[nt] = wmma_bf16(a, b, oc[nt]);
        }
      }
#pragma unroll
      for (int g = 0; g < 8; ++g) {
        int rowg = row0 + g + 8 * lhi;
#pragma unroll
        for (int nt = 0; nt < 8; ++nt)
          os[rowg * QS_LD + nt * 16 + lrow] = (bf16t)oc[nt][g];
      }
    }

    // ===== GEMM3: out = o @ w_out + b_out ; residual + LayerNorm =========
    {
      v8f dc[4] = {};
#pragma unroll
      for (int s = 0; s < 4; ++s) {       // K = hidden
        v16bf a = load16_gap(os + (row0 + lrow) * QS_LD + s * 32 + lhi * 8);
#pragma unroll
        for (int nt = 0; nt < 4; ++nt) {
          v16bf b = load16_seq(Wo + (s * 32 + lhi * 8) * EMBED + nt * 16 + lrow, EMBED);
          dc[nt] = wmma_bf16(a, b, dc[nt]);
        }
      }
      float y[4][8];
#pragma unroll
      for (int nt = 0; nt < 4; ++nt) {
        int col = nt * 16 + lrow;
        float bias = bo[col];
#pragma unroll
        for (int g = 0; g < 8; ++g) {
          int rowg = row0 + g + 8 * lhi;
          y[nt][g] = dc[nt][g] + bias + xs[rowg * XS_LD + col];  // +residual
        }
      }
#pragma unroll
      for (int g = 0; g < 8; ++g) {
        float sum = 0.f;
#pragma unroll
        for (int nt = 0; nt < 4; ++nt) sum += y[nt][g];
        sum = redsum16(sum);
        float mu = sum * (1.0f / EMBED);
        float var = 0.f;
#pragma unroll
        for (int nt = 0; nt < 4; ++nt) { float d = y[nt][g] - mu; var += d * d; }
        var = redsum16(var) * (1.0f / EMBED);
        float rstd = rsqrtf(var + LN_EPS);
        int rowg = row0 + g + 8 * lhi;
#pragma unroll
        for (int nt = 0; nt < 4; ++nt) {
          int col = nt * 16 + lrow;
          float o = (y[nt][g] - mu) * rstd * lg[col] + lb[col];
          xs[rowg * XS_LD + col] = o;
          xb[rowg * XB_LD + col] = (bf16t)o;
        }
      }
    }
    __syncthreads();  // guard qs/kT/vs reuse across layers
  }

  // ---- write result ----------------------------------------------------
  float* Op = out + bidx * (size_t)(NNEI * EMBED);
#pragma unroll 4
  for (int i = tid; i < NNEI * EMBED; i += 256) {
    int rI = i >> 6, cI = i & 63;
    Op[i] = xs[rI * XS_LD + cI];
  }
}

// ---------------------------------------------------------------------------
extern "C" void kernel_launch(void* const* d_in, const int* in_sizes, int n_in,
                              void* d_out, int out_size, void* d_ws, size_t ws_size,
                              hipStream_t stream) {
  (void)in_sizes; (void)n_in; (void)out_size; (void)ws_size;
  const float*         G     = (const float*)d_in[0];
  const unsigned char* nmask = (const unsigned char*)d_in[1];  // jnp bool (1B)
  const float*         rvec  = (const float*)d_in[2];
  const float*         sw    = (const float*)d_in[3];
  const float*         in_w  = (const float*)d_in[4];
  const float*         in_b  = (const float*)d_in[5];
  const float*         out_w = (const float*)d_in[6];
  const float*         out_b = (const float*)d_in[7];
  const float*         ln_g  = (const float*)d_in[8];
  const float*         ln_b  = (const float*)d_in[9];

  bf16t* wi = (bf16t*)d_ws;                            // 2*64*384 bf16
  bf16t* wo = wi + NLAYER * EMBED * 3 * HIDDEN;        // 2*128*64 bf16

  convert_weights_kernel<<<192, 256, 0, stream>>>(in_w, out_w, wi, wo);

  attn_fused_kernel<<<NB, 256, SMEM_BYTES, stream>>>(
      G, nmask, rvec, sw, in_b, out_b, ln_g, ln_b, wi, wo, (float*)d_out);
}
